// PlasVarDyn_18597208391867
// MI455X (gfx1250) — matrix operand order repaired and checked by reading, not compiled, not run on hardware
//
#include <hip/hip_runtime.h>
#include <hip/hip_bf16.h>
#include <math.h>

// Problem constants (from reference)
#define D_DIM   1024
#define N_BATCH 64
#define T_STEPS 256
#define B_TOK   (N_BATCH * T_STEPS)   // 16384 tokens
#define H_HEADS 8
#define HD_DIM  128

#define GLOBAL_AS __attribute__((address_space(1)))
#define LDS_AS    __attribute__((address_space(3)))

typedef __attribute__((ext_vector_type(16))) __bf16 v16bf;
typedef __attribute__((ext_vector_type(8)))  __bf16 v8bf;
typedef __attribute__((ext_vector_type(8)))  float  v8f;

typedef const LDS_AS    __bf16* lds_bf_p;
typedef const GLOBAL_AS __bf16* glob_bf_p;

// ----------------------------------------------------------------------------
// Weight pre-pack: fp32 row-major W[k][n] -> bf16, reordered so that each
// lane's 16-element B-fragment (V_WMMA_F32_16X16X32_BF16, 32x16 B tile) is
// contiguous in memory.  Fragment layout (contiguous k order used here, 16-bit B):
//   lanes 0-15 : N = lane,     K = kt*32 + 0..15
//   lanes 16-31: N = lane-16,  K = kt*32 + 16..31
// Packed index: (((ntile*32 + kt)*32 + lane)*16 + j)
//            <-> W[(kt*32 + (lane&16) + j)][ntile*16 + (lane&15)]
// ----------------------------------------------------------------------------
struct WeightPtrs { const float* w[7]; };

__global__ void convert_pack_weights(WeightPtrs src, __bf16* __restrict__ dst) {
  int idx = blockIdx.x * blockDim.x + threadIdx.x;  // 7 * 1,048,576 total
  int j     =  idx        & 15;
  int lane  = (idx >> 4)  & 31;
  int kt    = (idx >> 9)  & 31;
  int ntile = (idx >> 14) & 63;
  int mat   =  idx >> 20;
  int k = kt * 32 + (lane & 16) + j;
  int n = ntile * 16 + (lane & 15);
  dst[idx] = (__bf16)src.w[mat][(size_t)k * D_DIM + n];
}

// ----------------------------------------------------------------------------
// One 16x16 output tile of  X[16,1024] @ W[1024, ntile*16 .. +16), K swept in
// 32 steps of 32 (bf16), fp32 accumulate, TWO accumulators to break the
// WMMA C->D dependency chain.  X from LDS (ds_load_b128), W from the packed
// global layout (global_load_b128) -- address spaces explicit so codegen
// does NOT fall back to flat.
// A-fragment (16-bit A 16x32): lanes 0-15 hold K 0..15,
// lanes 16-31 hold K 16..31, M = lane&15.
// ----------------------------------------------------------------------------
__device__ inline v8f wmma_tile(lds_bf_p Xlds, int ldx,
                                glob_bf_p Wpack, int ntile, int lane) {
  v8f acc0 = {}, acc1 = {};
  const int m    = lane & 15;
  const int koff = (lane & 16);               // 0 or 16
  glob_bf_p wp    = Wpack + ((size_t)(ntile * 32) * 32 + lane) * 16;
  lds_bf_p  xbase = Xlds + (size_t)m * ldx + koff;
  for (int kt = 0; kt < 32; kt += 2) {
    v8bf a0lo = *(const LDS_AS v8bf*)(xbase + kt * 32);
    v8bf a0hi = *(const LDS_AS v8bf*)(xbase + kt * 32 + 8);
    v8bf a1lo = *(const LDS_AS v8bf*)(xbase + kt * 32 + 32);
    v8bf a1hi = *(const LDS_AS v8bf*)(xbase + kt * 32 + 40);
    v16bf b0 = *(const GLOBAL_AS v16bf*)(wp);
    v16bf b1 = *(const GLOBAL_AS v16bf*)(wp + 512);
    v16bf a0 = __builtin_shufflevector(a0lo, a0hi,
               0,1,2,3,4,5,6,7,8,9,10,11,12,13,14,15);
    v16bf a1 = __builtin_shufflevector(a1lo, a1hi,
               0,1,2,3,4,5,6,7,8,9,10,11,12,13,14,15);
    acc0 = __builtin_amdgcn_wmma_f32_16x16x32_bf16(
               false, a0, false, b0, (short)0, acc0, false, false);
    acc1 = __builtin_amdgcn_wmma_f32_16x16x32_bf16(
               false, a1, false, b1, (short)0, acc1, false, false);
    wp += 1024;                               // 2 kt-chunks * 512 bf16
  }
  return acc0 + acc1;
}

// ----------------------------------------------------------------------------
// PlasVar layer, fully fused per 16-token tile:
//   Q/K/V = X@W + b  (WMMA, bf16 in, fp32 acc)   -> LDS
//   per-token  QK[8,8] = Q K^T ;  attn = QK @ V * 1/sqrt(D) ; x += attn
// One workgroup (8 waves) per 16-token tile; 1024 workgroups.
// ----------------------------------------------------------------------------
__global__ void __launch_bounds__(256)
var_layer(const float* __restrict__ Xin, float* __restrict__ Xout,
          const __bf16* __restrict__ Wq_, const __bf16* __restrict__ Wk_,
          const __bf16* __restrict__ Wv_,
          const float* __restrict__ bq, const float* __restrict__ bk,
          const float* __restrict__ bv)
{
  __shared__ __bf16 Xbf[16][1032];        // padded row stride (16B aligned)
  __shared__ __bf16 QKV[3][16][1032];
  __shared__ float  QKs[16][64];

  const int tid = threadIdx.x, lane = tid & 31, wave = tid >> 5;
  const int rowbase = blockIdx.x * 16;

  // stage + convert X tile to bf16 LDS
  for (int i = tid; i < 16 * D_DIM; i += 256) {
    int r = i >> 10, c = i & 1023;
    Xbf[r][c] = (__bf16)Xin[(size_t)(rowbase + r) * D_DIM + c];
  }
  __syncthreads();

  glob_bf_p Wmat[3] = {(glob_bf_p)Wq_, (glob_bf_p)Wk_, (glob_bf_p)Wv_};
  const float* Bias[3] = {bq, bk, bv};
  const int mbase = (lane & 16) >> 1;     // C/D layout: lanes16-31 -> rows 8..15
  const int ncol  = lane & 15;

  // 3 matrices x 64 column tiles = 192 WMMA jobs over 8 waves
  for (int job = wave; job < 192; job += 8) {
    int mat = job >> 6, ntile = job & 63;
    v8f acc = wmma_tile((lds_bf_p)&Xbf[0][0], 1032, Wmat[mat], ntile, lane);
    float bb = Bias[mat][ntile * 16 + ncol];
#pragma unroll
    for (int i = 0; i < 8; ++i)
      QKV[mat][mbase + i][ntile * 16 + ncol] = (__bf16)(acc[i] + bb);
  }
  __syncthreads();

  // local per-token attention: 16 threads per token
  const int t = tid >> 4, part = tid & 15;
  for (int e = part * 4; e < part * 4 + 4; ++e) {
    int h = e >> 3, g = e & 7;
    float s = 0.f;
    for (int d = 0; d < HD_DIM; ++d)
      s += (float)QKV[0][t][h * 128 + d] * (float)QKV[1][t][g * 128 + d];
    QKs[t][e] = s;
  }
  __syncthreads();

  const float scale = 0.03125f;           // 1/sqrt(1024)
  for (int j = 0; j < 64; ++j) {
    int c = part * 64 + j;
    int h = c >> 7, dl = c & 127;
    float s = 0.f;
#pragma unroll
    for (int g = 0; g < 8; ++g)
      s += QKs[t][h * 8 + g] * (float)QKV[2][t][g * 128 + dl];
    size_t gi = (size_t)(rowbase + t) * D_DIM + c;
    Xout[gi] = Xin[gi] + s * scale;
  }
}

// ----------------------------------------------------------------------------
// A = X @ Wa + ba  (hoisted out of the scan).  Grid: 1024 row tiles x 8 column
// groups; each wave owns one 16x16 tile.
// ----------------------------------------------------------------------------
__global__ void __launch_bounds__(256)
gemm_bias(const float* __restrict__ Xin, const __bf16* __restrict__ Wp_,
          const float* __restrict__ bias, float* __restrict__ Out)
{
  __shared__ __bf16 Xbf[16][1032];
  const int tid = threadIdx.x, lane = tid & 31, wave = tid >> 5;
  const int rowtile = blockIdx.x >> 3, cg = blockIdx.x & 7;
  const int rowbase = rowtile * 16;

  for (int i = tid; i < 16 * D_DIM; i += 256) {
    int r = i >> 10, c = i & 1023;
    Xbf[r][c] = (__bf16)Xin[(size_t)(rowbase + r) * D_DIM + c];
  }
  __syncthreads();

  int ntile = cg * 8 + wave;
  v8f acc = wmma_tile((lds_bf_p)&Xbf[0][0], 1032, (glob_bf_p)Wp_, ntile, lane);
  int mbase = (lane & 16) >> 1, ncol = lane & 15;
  int n = ntile * 16 + ncol;
  float bb = bias[n];
#pragma unroll
  for (int i = 0; i < 8; ++i)
    Out[(size_t)(rowbase + mbase + i) * D_DIM + n] = acc[i] + bb;
}

// ----------------------------------------------------------------------------
// One PlasDyn scan step:
//   h_new = (h@Wm+bm)*sig(h@Wmg+bmg) + A_t * sig(h@Wag+bag)
// Grid: 4 row tiles (64 rows) x 8 column groups (128 cols) = 32 WGs.
// The A_t tile (16x128 fp32, 8KB) is prefetched Global->LDS with the CDNA5
// async path (global_load_async_to_lds_b128, ASYNCcnt) fully overlapped with
// the three gate GEMMs; we s_wait_asynccnt only right before the combine.
// A is stored in the x layout: A[(n*T + t)*D + d].
// ----------------------------------------------------------------------------
__global__ void __launch_bounds__(256)
dyn_step(const float* __restrict__ Hin, float* __restrict__ Hout,
         const float* __restrict__ A, int t,
         const __bf16* __restrict__ Wm_, const __bf16* __restrict__ Wg_,
         const __bf16* __restrict__ Wa_,
         const float* __restrict__ bm, const float* __restrict__ bg,
         const float* __restrict__ ba)
{
  __shared__ __bf16 Hbf[16][1032];
  __shared__ float  R[3][16][128];
  __shared__ float  Alds[16][128];

  const int tid = threadIdx.x, lane = tid & 31, wave = tid >> 5;
  const int mtile = blockIdx.x >> 3, cg = blockIdx.x & 7;
  const int rowbase = mtile * 16;
  const int colbase = cg * 128;

  // kick off async Global->LDS staging of the A_t tile (2 x 16B per thread)
  for (int c = tid; c < 512; c += 256) {      // 512 chunks of 16 bytes
    int r  = c >> 5;                          // 32 chunks per 128-float row
    int cc = c & 31;
    const float* gsrc =
        &A[((size_t)(rowbase + r) * T_STEPS + t) * D_DIM + colbase + cc * 4];
    unsigned long long ga = (unsigned long long)(uintptr_t)gsrc;
    unsigned int loff =
        (unsigned int)(uintptr_t)(LDS_AS float*)&Alds[r][cc * 4];
    asm volatile("global_load_async_to_lds_b128 %0, %1, off"
                 :: "v"(loff), "v"(ga) : "memory");
  }

  // stage + convert h rows to bf16 LDS
  for (int i = tid; i < 16 * D_DIM; i += 256) {
    int r = i >> 10, c = i & 1023;
    Hbf[r][c] = (__bf16)Hin[(size_t)(rowbase + r) * D_DIM + c];
  }
  __syncthreads();

  glob_bf_p Wmat[3] = {(glob_bf_p)Wm_, (glob_bf_p)Wg_, (glob_bf_p)Wa_};
  const float* Bias[3] = {bm, bg, ba};
  const int mbase = (lane & 16) >> 1, ncol = lane & 15;

  for (int job = wave; job < 24; job += 8) {
    int mat = job >> 3, nt = job & 7;
    int ntile = cg * 8 + nt;
    v8f acc = wmma_tile((lds_bf_p)&Hbf[0][0], 1032, Wmat[mat], ntile, lane);
    float bb = Bias[mat][ntile * 16 + ncol];
#pragma unroll
    for (int i = 0; i < 8; ++i)
      R[mat][mbase + i][nt * 16 + ncol] = acc[i] + bb;
  }

  // A_t tile must be resident in LDS before the combine
  asm volatile("s_wait_asynccnt 0x0" ::: "memory");
  __syncthreads();

  for (int e = tid * 8; e < tid * 8 + 8; ++e) {
    int r = e >> 7, c = e & 127;
    int nrow = rowbase + r;
    float m_ = R[0][r][c];
    float g_ = 1.f / (1.f + __expf(-R[1][r][c]));
    float a_ = 1.f / (1.f + __expf(-R[2][r][c]));
    Hout[(size_t)nrow * D_DIM + colbase + c] = m_ * g_ + Alds[r][c] * a_;
  }
}

__global__ void zero_f32(float* __restrict__ p, int n) {
  int i = blockIdx.x * blockDim.x + threadIdx.x;
  if (i < n) p[i] = 0.f;
}

// ----------------------------------------------------------------------------
extern "C" void kernel_launch(void* const* d_in, const int* in_sizes, int n_in,
                              void* d_out, int out_size, void* d_ws, size_t ws_size,
                              hipStream_t stream) {
  const float* V   = (const float*)d_in[0];
  const float* Wq  = (const float*)d_in[1];  const float* bq  = (const float*)d_in[2];
  const float* Wk  = (const float*)d_in[3];  const float* bk  = (const float*)d_in[4];
  const float* Wv  = (const float*)d_in[5];  const float* bv  = (const float*)d_in[6];
  const float* Wm  = (const float*)d_in[7];  const float* bm  = (const float*)d_in[8];
  const float* Wmg = (const float*)d_in[9];  const float* bmg = (const float*)d_in[10];
  const float* Wa  = (const float*)d_in[11]; const float* ba  = (const float*)d_in[12];
  const float* Wag = (const float*)d_in[13]; const float* bag = (const float*)d_in[14];

  char* ws = (char*)d_ws;
  size_t off = 0;
  float*  x0  = (float*)(ws + off);  off += (size_t)B_TOK * D_DIM * 4;      // 64 MB
  float*  x1  = (float*)(ws + off);  off += (size_t)B_TOK * D_DIM * 4;      // 64 MB, reused as A
  __bf16* wpk = (__bf16*)(ws + off); off += (size_t)7 * D_DIM * D_DIM * 2;  // 14 MB
  float*  h0  = (float*)(ws + off);  off += (size_t)N_BATCH * D_DIM * 4;
  float*  h1  = (float*)(ws + off);  off += (size_t)N_BATCH * D_DIM * 4;

  const size_t MAT = (size_t)D_DIM * D_DIM;
  __bf16* Wq_p  = wpk + 0 * MAT;
  __bf16* Wk_p  = wpk + 1 * MAT;
  __bf16* Wv_p  = wpk + 2 * MAT;
  __bf16* Wm_p  = wpk + 3 * MAT;
  __bf16* Wmg_p = wpk + 4 * MAT;
  __bf16* Wa_p  = wpk + 5 * MAT;
  __bf16* Wag_p = wpk + 6 * MAT;

  // 1) pack weights to WMMA-fragment-order bf16
  WeightPtrs wp;
  wp.w[0] = Wq; wp.w[1] = Wk; wp.w[2] = Wv;
  wp.w[3] = Wm; wp.w[4] = Wmg; wp.w[5] = Wa; wp.w[6] = Wag;
  {
    int total = 7 * (int)MAT;
    convert_pack_weights<<<total / 256, 256, 0, stream>>>(wp, wpk);
  }

  // 2) PlasVar: two residual linear-attention layers (V -> x1 -> x0)
  var_layer<<<B_TOK / 16, 256, 0, stream>>>(V,  x1, Wq_p, Wk_p, Wv_p, bq, bk, bv);
  var_layer<<<B_TOK / 16, 256, 0, stream>>>(x1, x0, Wq_p, Wk_p, Wv_p, bq, bk, bv);

  // 3) A = x @ Wa + ba (hoisted out of the scan); stored into x1's space
  float* A = x1;
  gemm_bias<<<(B_TOK / 16) * 8, 256, 0, stream>>>(x0, Wa_p, ba, A);

  // 4) sequential GLU scan over T=256 steps, h ping-pong
  zero_f32<<<(N_BATCH * D_DIM + 255) / 256, 256, 0, stream>>>(h0, N_BATCH * D_DIM);
  for (int t = 0; t < T_STEPS; ++t) {
    const float* hin = (t & 1) ? h1 : h0;
    float*       hou = (t & 1) ? h0 : h1;
    dyn_step<<<32, 256, 0, stream>>>(hin, hou, A, t,
                                     Wm_p, Wmg_p, Wag_p, bm, bmg, bag);
  }
  // T even -> final state lands back in h0
  hipMemcpyAsync(d_out, h0, (size_t)N_BATCH * D_DIM * 4,
                 hipMemcpyDeviceToDevice, stream);
}
